// MixtralSparseMoeBlock_62079457296768
// MI455X (gfx1250) — matrix operand order repaired and checked by reading, not compiled, not run on hardware
//
#include <hip/hip_runtime.h>

// ---------------- problem constants ----------------
constexpr int T    = 2048;    // B*S tokens
constexpr int Hd   = 1024;    // hidden
constexpr int NE   = 8;       // experts
constexpr int Id   = 3584;    // intermediate
constexpr int TM   = 32;      // tokens per FFN tile
constexpr int CI   = 128;     // I-chunk per iteration (8 waves x 16 channels)
constexpr int GLD  = CI + 8;  // padded LDS stride for g tile (halves)
constexpr int XLD  = Hd + 8;  // padded LDS stride for x tile (halves)

// ---------------- vector types ----------------
typedef __attribute__((ext_vector_type(16))) __bf16         v16bf;
typedef __attribute__((ext_vector_type(8)))  float          v8f;
typedef __attribute__((ext_vector_type(4)))  unsigned int   v4u;   // 16B chunk
typedef __attribute__((ext_vector_type(4)))  unsigned short v4h;

// fp32 -> bf16 round-to-nearest-even (raw ushort)
__device__ __forceinline__ unsigned short f2bf(float f) {
    unsigned int u = __float_as_uint(f);
    u += 0x7FFFu + ((u >> 16) & 1u);
    return (unsigned short)(u >> 16);
}

// B fragment: lane holds 16 consecutive bf16 (column n, K = hi*16 .. +15)
__device__ __forceinline__ v16bf ld_b_frag(const unsigned short* p) {
    v16bf r;
    reinterpret_cast<v4u*>(&r)[0] = *reinterpret_cast<const v4u*>(p);
    reinterpret_cast<v4u*>(&r)[1] = *reinterpret_cast<const v4u*>(p + 8);
    return r;
}

// A fragment: lane holds row m; K = k0+hi*16 .. +15 (contiguous)
__device__ __forceinline__ v16bf ld_a_frag(const unsigned short* row, int k0, int hi) {
    v16bf r;
    reinterpret_cast<v4u*>(&r)[0] = *reinterpret_cast<const v4u*>(row + k0 + hi * 16);
    reinterpret_cast<v4u*>(&r)[1] = *reinterpret_cast<const v4u*>(row + k0 + hi * 16 + 8);
    return r;
}

// ---------------- K0: fp32 -> bf16 conversion ----------------
__global__ __launch_bounds__(256) void cvt_bf16_kernel(const float* __restrict__ s,
                                                       unsigned short* __restrict__ d,
                                                       int n4) {
    int i = blockIdx.x * 256 + threadIdx.x;
    if (i >= n4) return;
    float4 v = reinterpret_cast<const float4*>(s)[i];
    v4h o = { f2bf(v.x), f2bf(v.y), f2bf(v.z), f2bf(v.w) };
    reinterpret_cast<v4h*>(d)[i] = o;
}

// ---------------- K1: router (fp32) ----------------
__global__ __launch_bounds__(256) void router_kernel(const float* __restrict__ x,
                                                     const float* __restrict__ gw,
                                                     int*   __restrict__ counts,
                                                     int*   __restrict__ toks,
                                                     float* __restrict__ wts) {
    __shared__ float sGW[NE * Hd];
    for (int i = threadIdx.x; i < NE * Hd; i += 256) sGW[i] = gw[i];
    __syncthreads();

    const int wave = threadIdx.x >> 5;
    const int lane = threadIdx.x & 31;
    const int t = blockIdx.x * 8 + wave;
    const float* xr = x + (size_t)t * Hd;

    float acc[NE];
#pragma unroll
    for (int e = 0; e < NE; e++) acc[e] = 0.f;
    for (int h = lane; h < Hd; h += 32) {
        float xv = xr[h];
#pragma unroll
        for (int e = 0; e < NE; e++) acc[e] = __builtin_fmaf(xv, sGW[e * Hd + h], acc[e]);
    }
#pragma unroll
    for (int off = 16; off > 0; off >>= 1) {
#pragma unroll
        for (int e = 0; e < NE; e++) acc[e] += __shfl_xor(acc[e], off, 32);
    }
    if (lane == 0) {
        float mx = acc[0];
#pragma unroll
        for (int e = 1; e < NE; e++) mx = fmaxf(mx, acc[e]);
        float p[NE];
#pragma unroll
        for (int e = 0; e < NE; e++) p[e] = __expf(acc[e] - mx);
        // softmax denominator cancels in top-2 renormalization.
        int b0 = 0;
#pragma unroll
        for (int e = 1; e < NE; e++) if (p[e] > p[b0]) b0 = e;
        int b1 = (b0 == 0) ? 1 : 0;
#pragma unroll
        for (int e = 0; e < NE; e++) if (e != b0 && p[e] > p[b1]) b1 = e;
        float inv = 1.f / (p[b0] + p[b1]);
        float w0 = p[b0] * inv, w1 = p[b1] * inv;
        int q0 = atomicAdd(&counts[b0], 1);
        toks[b0 * T + q0] = t;  wts[b0 * T + q0] = w0;
        int q1 = atomicAdd(&counts[b1], 1);
        toks[b1 * T + q1] = t;  wts[b1 * T + q1] = w1;
    }
}

// ---------------- K2: grouped fused SwiGLU FFN (WMMA bf16) ----------------
// Block = 32 gathered tokens x 1 expert, 8 waves. x tile staged to LDS with
// async-to-LDS copies. Per I-chunk of 128:
//   Phase 1: wave w owns I-slice [i0+16w, +16): 32 K-steps x 4 WMMAs
//            (2 token subtiles x {w1,w3}); every w1/w3 fragment loaded by
//            exactly ONE wave (no duplicate L2 traffic). g -> LDS (bf16).
//   Phase 2: wave w owns H-slice [128w, +128) (8 n-tiles): each w2 fragment
//            loaded once, feeds 2 WMMAs (both token subtiles). 128 acc VGPRs.
// Writeout: gate-weighted global_atomic_add_f32.
__global__ __launch_bounds__(256) void moe_ffn_kernel(const unsigned short* __restrict__ xb,
                                                      const unsigned short* __restrict__ w1b,
                                                      const unsigned short* __restrict__ w3b,
                                                      const unsigned short* __restrict__ w2b,
                                                      const int*   __restrict__ counts,
                                                      const int*   __restrict__ toks,
                                                      const float* __restrict__ wts,
                                                      float* __restrict__ out) {
    const int e     = blockIdx.y;
    const int count = counts[e];
    const int base  = blockIdx.x * TM;
    if (base >= count) return;   // uniform across block; safe before barriers

    __shared__ __align__(16) unsigned short sX[TM * XLD]; // 66 KB gathered x tile
    __shared__ __align__(16) unsigned short sG[TM * GLD]; // 8.7 KB g tile
    __shared__ int   sTok[TM];
    __shared__ float sW[TM];

    if (threadIdx.x < TM) {
        int idx = base + threadIdx.x;
        if (idx < count) { sTok[threadIdx.x] = toks[e * T + idx]; sW[threadIdx.x] = wts[e * T + idx]; }
        else             { sTok[threadIdx.x] = toks[e * T + base]; sW[threadIdx.x] = 0.f; } // pad: weight 0
    }
    __syncthreads();

    // ---- stage gathered x rows into LDS with async global->LDS copies ----
    {
        const int r   = threadIdx.x >> 3;          // row 0..31
        const int sub = threadIdx.x & 7;           // 8 threads per row
        const unsigned short* src = xb + (size_t)sTok[r] * Hd + sub * 128;
        unsigned short* dst = &sX[r * XLD + sub * 128];
#pragma unroll
        for (int c = 0; c < 16; c++) {             // 16 x 16B per thread
            unsigned lds = (unsigned)(uintptr_t)(dst + c * 8);
            const unsigned short* g = src + c * 8;
            asm volatile("global_load_async_to_lds_b128 %0, %1, off"
                         :: "v"(lds), "v"(g) : "memory");
        }
        asm volatile("s_wait_asynccnt 0" ::: "memory");
    }
    __syncthreads();

    const int wave = threadIdx.x >> 5;
    const int lane = threadIdx.x & 31;
    const int fn   = lane & 15;   // fragment column / row-in-subtile
    const int hi   = lane >> 4;   // half-wave selector

    v8f acc[16] = {};             // phase-2: [mi][nt] -> acc[mi*8+nt]

    for (int i0 = 0; i0 < Id; i0 += CI) {
        // -------- phase 1: h1/h3 slices, K = H, both token subtiles --------
        v8f a1_0 = {}, a3_0 = {}, a1_1 = {}, a3_1 = {};
        const unsigned short* w1r = w1b + ((size_t)e * Id + i0 + wave * 16 + fn) * Hd;
        const unsigned short* w3r = w3b + ((size_t)e * Id + i0 + wave * 16 + fn) * Hd;
        if (i0 + CI < Id) {       // prefetch next chunk's B rows into cache
            __builtin_prefetch(w1r + (size_t)CI * Hd, 0, 3);
            __builtin_prefetch(w3r + (size_t)CI * Hd, 0, 3);
        }
        const unsigned short* x0 = &sX[(0 * 16 + fn) * XLD];
        const unsigned short* x1 = &sX[(1 * 16 + fn) * XLD];
        for (int k0 = 0; k0 < Hd; k0 += 32) {
            v16bf A0 = ld_a_frag(x0, k0, hi);
            v16bf A1 = ld_a_frag(x1, k0, hi);
            v16bf B1 = ld_b_frag(w1r + k0 + hi * 16);
            v16bf B3 = ld_b_frag(w3r + k0 + hi * 16);
            a1_0 = __builtin_amdgcn_wmma_f32_16x16x32_bf16(false, A0, false, B1, (short)0, a1_0, false, false);
            a3_0 = __builtin_amdgcn_wmma_f32_16x16x32_bf16(false, A0, false, B3, (short)0, a3_0, false, false);
            a1_1 = __builtin_amdgcn_wmma_f32_16x16x32_bf16(false, A1, false, B1, (short)0, a1_1, false, false);
            a3_1 = __builtin_amdgcn_wmma_f32_16x16x32_bf16(false, A1, false, B3, (short)0, a3_1, false, false);
        }
        __syncthreads();          // WAR: prior phase-2 reads of sG complete
#pragma unroll
        for (int r = 0; r < 8; r++) {
            float h1 = a1_0[r], h3 = a3_0[r];
            float g0 = (h1 / (1.f + __expf(-h1))) * h3;          // silu(h1)*h3
            sG[(hi * 8 + r) * GLD + wave * 16 + fn] = f2bf(g0);
            float j1 = a1_1[r], j3 = a3_1[r];
            float g1 = (j1 / (1.f + __expf(-j1))) * j3;
            sG[(16 + hi * 8 + r) * GLD + wave * 16 + fn] = f2bf(g1);
        }
        __syncthreads();          // g visible to all waves

        // -------- phase 2: y += g @ W2^T, K = CI; w2 fragment -> 2 WMMAs ----
        const unsigned short* g0 = &sG[(0 * 16 + fn) * GLD];
        const unsigned short* g1 = &sG[(1 * 16 + fn) * GLD];
#pragma unroll
        for (int kk = 0; kk < CI; kk += 32) {
            v16bf A0 = ld_a_frag(g0, kk, hi);
            v16bf A1 = ld_a_frag(g1, kk, hi);
#pragma unroll
            for (int nt = 0; nt < 8; nt++) {
                const unsigned short* w2r =
                    w2b + ((size_t)e * Hd + wave * 128 + nt * 16 + fn) * Id + i0 + kk + hi * 16;
                v16bf B = ld_b_frag(w2r);
                acc[nt]     = __builtin_amdgcn_wmma_f32_16x16x32_bf16(false, A0, false, B, (short)0, acc[nt],     false, false);
                acc[8 + nt] = __builtin_amdgcn_wmma_f32_16x16x32_bf16(false, A1, false, B, (short)0, acc[8 + nt], false, false);
            }
        }
    }

    // -------- writeout: gate-weighted scatter-add --------
#pragma unroll
    for (int mi = 0; mi < 2; mi++) {
#pragma unroll
        for (int nt = 0; nt < 8; nt++) {
            int h0 = wave * 128 + nt * 16 + fn;
#pragma unroll
            for (int r = 0; r < 8; r++) {
                int m = mi * 16 + hi * 8 + r;
                atomicAdd(&out[(size_t)sTok[m] * Hd + h0], sW[m] * acc[mi * 8 + nt][r]);
            }
        }
    }
}

// ---------------- host launcher ----------------
extern "C" void kernel_launch(void* const* d_in, const int* in_sizes, int n_in,
                              void* d_out, int out_size, void* d_ws, size_t ws_size,
                              hipStream_t stream) {
    const float* x  = (const float*)d_in[0];   // [T,H]
    const float* gw = (const float*)d_in[1];   // [E,H]
    const float* w1 = (const float*)d_in[2];   // [E,I,H]
    const float* w2 = (const float*)d_in[3];   // [E,H,I]
    const float* w3 = (const float*)d_in[4];   // [E,I,H]
    float* out = (float*)d_out;

    // workspace layout (256B aligned): counts | tok lists | wt lists | bf16 x,w1,w3,w2
    char* ws = (char*)d_ws;
    constexpr size_t OFF_CNT = 0;
    constexpr size_t OFF_TOK = 256;
    constexpr size_t OFF_WT  = OFF_TOK + (size_t)NE * T * 4;
    constexpr size_t OFF_XB  = OFF_WT  + (size_t)NE * T * 4;
    constexpr size_t SZ_W    = (size_t)NE * Id * Hd * 2;
    constexpr size_t OFF_W1B = OFF_XB  + (size_t)T * Hd * 2;
    constexpr size_t OFF_W3B = OFF_W1B + SZ_W;
    constexpr size_t OFF_W2B = OFF_W3B + SZ_W;

    int*            counts = (int*)(ws + OFF_CNT);
    int*            toks   = (int*)(ws + OFF_TOK);
    float*          wts    = (float*)(ws + OFF_WT);
    unsigned short* xb     = (unsigned short*)(ws + OFF_XB);
    unsigned short* w1b    = (unsigned short*)(ws + OFF_W1B);
    unsigned short* w3b    = (unsigned short*)(ws + OFF_W3B);
    unsigned short* w2b    = (unsigned short*)(ws + OFF_W2B);

    hipMemsetAsync(counts, 0, NE * sizeof(int), stream);
    hipMemsetAsync(out, 0, (size_t)T * Hd * sizeof(float), stream);

    // bf16 conversions (weights become L2-resident: 176MB < 192MB L2)
    {
        int n4 = (T * Hd) / 4;
        cvt_bf16_kernel<<<(n4 + 255) / 256, 256, 0, stream>>>(x, xb, n4);
    }
    {
        int n4 = (NE * Id * Hd) / 4;
        cvt_bf16_kernel<<<(n4 + 255) / 256, 256, 0, stream>>>(w1, w1b, n4);
        cvt_bf16_kernel<<<(n4 + 255) / 256, 256, 0, stream>>>(w3, w3b, n4);
        cvt_bf16_kernel<<<(n4 + 255) / 256, 256, 0, stream>>>(w2, w2b, n4);
    }

    // router: one wave per token
    router_kernel<<<T / 8, 256, 0, stream>>>(x, gw, counts, toks, wts);

    // grouped FFN: (m-tiles, experts); surplus tiles exit on count check
    dim3 grid(T / TM, NE);
    moe_ffn_kernel<<<grid, 256, 0, stream>>>(xb, w1b, w3b, w2b, counts, toks, wts, out);
}
